// DecoderBlock_88158498718113
// MI455X (gfx1250) — compile-verified, matrix operand order checked by reading, not run on hardware
//
#include <hip/hip_runtime.h>
#include <hip/hip_bf16.h>
#include <math.h>

// ---------------- problem constants ----------------
#define DM    512      // d_model
#define DIN   1024     // d_inner
#define NS    16       // d_state
#define DTR   32       // dt_rank
#define NB    32       // batch
#define TLOW  512
#define TT    1024
#define MFULL (NB * TT)      // 32768 rows (b,t)
#define NXP   (DTR + 2 * NS) // 80

typedef __attribute__((ext_vector_type(16))) _Float16 v16h;
typedef __attribute__((ext_vector_type(8)))  _Float16 v8h;
typedef __attribute__((ext_vector_type(8)))  float    v8f;

// ---------------- CDNA5 wait helpers ----------------
#if __has_builtin(__builtin_amdgcn_s_wait_asynccnt)
#define WAIT_ASYNC(n) __builtin_amdgcn_s_wait_asynccnt(n)
#else
#define WAIT_ASYNC(n) asm volatile("s_wait_asynccnt %0" :: "i"(n) : "memory")
#endif

// Async DMA one 16B chunk: global -> LDS (per-lane), tracked by ASYNCcnt.
// GV mode: vdst = LDS byte offset (low 32 bits of generic shared pointer),
// vaddr = 64-bit global address, saddr = off.
__device__ __forceinline__ void async_copy_b128(void* lds_dst, const void* gsrc) {
  unsigned dst = (unsigned)(uintptr_t)lds_dst;
  unsigned long long src = (unsigned long long)(uintptr_t)gsrc;
  asm volatile("global_load_async_to_lds_b128 %0, %1, off"
               :: "v"(dst), "v"(src) : "memory");
}

// ---------------- WMMA helpers (CDNA5 wave32) ----------------
__device__ __forceinline__ v8f wmma_f32_f16(v16h a, v16h b, v8f c) {
  // D = A(16x32 f16) * B(32x16 f16) + C(16x16 f32)
  return __builtin_amdgcn_wmma_f32_16x16x32_f16(
      false, a, false, b, (short)0, c, false, false);
}

// Build a 16x32 f16 A/B fragment from a per-lane row pointer.
// ISA layout: lane L holds row (L&15); halves 0..7 = K 0..7 (+8 if L>=16),
// halves 8..15 = K 16..23 (+8 if L>=16). Caller pre-adds 8*(L>>4).
__device__ __forceinline__ v16h frag_rows(const _Float16* __restrict__ p) {
  v8h lo = *(const v8h*)(p);
  v8h hi = *(const v8h*)(p + 16);
  return __builtin_shufflevector(lo, hi, 0,1,2,3,4,5,6,7,8,9,10,11,12,13,14,15);
}

__device__ __forceinline__ v16h frag_load(const _Float16* __restrict__ base, int ld) {
  int lane = threadIdx.x & 31;
  return frag_rows(base + (size_t)(lane & 15) * ld + 8 * (lane >> 4));
}

// ---------------- small elementwise / precompute kernels ----------------
__global__ void cvt_f16_kernel(const float* __restrict__ src,
                               _Float16* __restrict__ dst, size_t n) {
  size_t i = (size_t)blockIdx.x * blockDim.x + threadIdx.x;
  if (i < n) dst[i] = (_Float16)src[i];
}

// Ct[k][o][cin] = sum_c up_w[cin][c][k] * fus_w[o][c]   (composite convT*1x1)
__global__ void build_composite_kernel(const float* __restrict__ up_w,
                                       const float* __restrict__ fus_w,
                                       _Float16* __restrict__ Ct) {
  int idx = blockIdx.x * blockDim.x + threadIdx.x;   // k*DM*DM + o*DM + cin
  if (idx >= 4 * DM * DM) return;
  int cin = idx & (DM - 1);
  int o   = (idx >> 9) & (DM - 1);
  int k   = idx >> 18;
  float s = 0.f;
  for (int c = 0; c < DM; ++c)
    s += up_w[((size_t)cin * DM + c) * 4 + k] * fus_w[(size_t)o * (2 * DM) + c];
  Ct[idx] = (_Float16)s;
}

// bias2[o] = fus_b[o] + sum_c up_b[c]*fus_w[o][c]
__global__ void build_bias2_kernel(const float* __restrict__ up_b,
                                   const float* __restrict__ fus_w,
                                   const float* __restrict__ fus_b,
                                   float* __restrict__ bias2) {
  int o = blockIdx.x * blockDim.x + threadIdx.x;
  if (o >= DM) return;
  float s = fus_b[o];
  for (int c = 0; c < DM; ++c) s += up_b[c] * fus_w[(size_t)o * (2 * DM) + c];
  bias2[o] = s;
}

// Fs[o][c] = fus_w[o][DM + c]  (skip-branch weights, f16)
__global__ void extract_fs_kernel(const float* __restrict__ fus_w,
                                  _Float16* __restrict__ Fs) {
  int i = blockIdx.x * blockDim.x + threadIdx.x;
  if (i >= DM * DM) return;
  int o = i >> 9, c = i & (DM - 1);
  Fs[i] = (_Float16)fus_w[(size_t)o * (2 * DM) + DM + c];
}

// ---------------- fused convtranspose + skip 1x1 (WMMA) ----------------
// parity 0: fused[2s]   = x[s]*C1 + x[s-1]*C3 + skip[2s]*Fs^T   + bias2
// parity 1: fused[2s+1] = x[s]*C2 + x[s+1]*C0 + skip[2s+1]*Fs^T + bias2
__global__ void fuse_kernel(const _Float16* __restrict__ x16,   // (NB,TLOW,DM)
                            const _Float16* __restrict__ sk16,  // (NB,TT,DM)
                            const _Float16* __restrict__ Ct,    // [4][DM][DM] (o,cin)
                            const _Float16* __restrict__ Fs,    // [DM][DM]   (o,c)
                            const float*    __restrict__ bias2,
                            float*          __restrict__ fused, // (NB,TT,DM)
                            int parity) {
  int wave = threadIdx.x >> 5, lane = threadIdx.x & 31;
  int n0 = (blockIdx.x * 4 + wave) * 16;
  int b  = blockIdx.y >> 5;                 // TLOW/16 == 32 tiles per batch
  int s0 = (blockIdx.y & 31) << 4;
  int r = lane & 15, hi = lane >> 4;

  int ka, kb, db;
  if (parity == 0) { ka = 1; kb = 3; db = -1; }
  else             { ka = 2; kb = 0; db = +1; }
  int sA = s0 + r;
  int sB = sA + db;
  bool vB = (unsigned)sB < (unsigned)TLOW;

  const _Float16* arow = x16 + ((size_t)b * TLOW + sA) * DM + 8 * hi;
  const _Float16* brow = x16 + ((size_t)b * TLOW + (vB ? sB : sA)) * DM + 8 * hi;
  const _Float16* srow = sk16 + ((size_t)b * TT + 2 * sA + parity) * DM + 8 * hi;
  const _Float16* B1 = Ct + (size_t)ka * DM * DM + (size_t)n0 * DM;
  const _Float16* B2 = Ct + (size_t)kb * DM * DM + (size_t)n0 * DM;
  const _Float16* B3 = Fs + (size_t)n0 * DM;

  v8f acc = {};
  for (int k = 0; k < DM; k += 32) {
    v16h a1 = frag_rows(arow + k);
    v16h a2 = {};
    if (vB) a2 = frag_rows(brow + k);
    v16h a3 = frag_rows(srow + k);
    v16h w1 = frag_load(B1 + k, DM);
    v16h w2 = frag_load(B2 + k, DM);
    v16h w3 = frag_load(B3 + k, DM);
    acc = wmma_f32_f16(a1, w1, acc);
    acc = wmma_f32_f16(a2, w2, acc);
    acc = wmma_f32_f16(a3, w3, acc);
  }
  int n = n0 + (lane & 15);
  float bn = bias2[n];
#pragma unroll
  for (int v = 0; v < 8; ++v) {
    int s = s0 + v + 8 * hi;
    int t = 2 * s + parity;
    fused[((size_t)b * TT + t) * DM + n] = acc[v] + bn;
  }
}

// ---------------- LayerNorm over DM ----------------
__global__ void ln_kernel(const float* __restrict__ fused,
                          const float* __restrict__ g, const float* __restrict__ be,
                          _Float16* __restrict__ hout) {
  __shared__ float s1[256], s2[256];
  int row = blockIdx.x, tid = threadIdx.x;
  const float* fr = fused + (size_t)row * DM;
  float a = fr[tid], b = fr[tid + 256];
  s1[tid] = a + b;
  s2[tid] = a * a + b * b;
  __syncthreads();
  for (int st = 128; st > 0; st >>= 1) {
    if (tid < st) { s1[tid] += s1[tid + st]; s2[tid] += s2[tid + st]; }
    __syncthreads();
  }
  float mu  = s1[0] * (1.f / DM);
  float var = s2[0] * (1.f / DM) - mu * mu;
  float inv = rsqrtf(var + 1e-5f);
  _Float16* ho = hout + (size_t)row * DM;
  ho[tid]       = (_Float16)((a - mu) * inv * g[tid]       + be[tid]);
  ho[tid + 256] = (_Float16)((b - mu) * inv * g[tid + 256] + be[tid + 256]);
}

// ---------------- WMMA GEMM: D = A * B^T (+bias)(+act) ----------------
// Block: 128 threads (4 waves). Block tile: 128M x 64N; wave tile: 32M x 64N.
// B^T tile (64 rows x 32 k-halfs) staged in LDS via async DMA, double-buffered.
// A: (M,K) f16 row-major, lda;  Bw: (N,K) f16 row-major (=B^T), ldb.
// ACT: 0 = identity, 1 = softplus.
template <int ACT, bool WRITE_F16>
__global__ __launch_bounds__(128)
void wmma_gemm_kernel(const _Float16* __restrict__ A, int lda,
                      const _Float16* __restrict__ Bw, int ldb,
                      const float* __restrict__ bias,
                      float* __restrict__ D, int ldd,
                      _Float16* __restrict__ D16, int ldd16,
                      int M, int N, int K) {
  constexpr int LDSB = 40;   // 32 halfs + 8 pad  (80B row, 16B aligned)
  __shared__ __align__(16) _Float16 Bs[2][64 * LDSB];

  int tid  = threadIdx.x;
  int wave = tid >> 5, lane = tid & 31;
  int nl = lane & 15, hi = lane >> 4;
  int n0 = blockIdx.x * 64;
  int m0 = blockIdx.y * 128 + wave * 32;

  const _Float16* Arow0 = A + (size_t)(m0 + nl) * lda + 8 * hi;
  const _Float16* Arow1 = Arow0 + (size_t)16 * lda;

  // stage one 64x32 B^T tile into LDS buffer `buf` (rows clamped: no OOB DMA)
  auto stage = [&](int buf, int ks) {
    for (int q = tid; q < 256; q += 128) {
      int row = q >> 2, part = q & 3;
      int gr = n0 + row;
      if (gr > N - 1) gr = N - 1;
      const _Float16* src = Bw + (size_t)gr * ldb + ks + part * 8;
      async_copy_b128(&Bs[buf][row * LDSB + part * 8], src);
    }
  };

  v8f acc[2][4] = {};
  int nk = K >> 5;
  stage(0, 0);
  for (int s = 0; s < nk; ++s) {
    int ks = s << 5;
    int cur = s & 1;
    if (s + 1 < nk) {
      stage(cur ^ 1, ks + 32);
      WAIT_ASYNC(2);          // current tile landed; next tile still in flight
    } else {
      WAIT_ASYNC(0);
    }
    __syncthreads();
    v16h a0 = frag_rows(Arow0 + ks);
    v16h a1 = frag_rows(Arow1 + ks);
#pragma unroll
    for (int j = 0; j < 4; ++j) {
      if (n0 + j * 16 < N) {  // block-uniform guard
        v16h bf = frag_rows(&Bs[cur][(j * 16 + nl) * LDSB + 8 * hi]);
        acc[0][j] = wmma_f32_f16(a0, bf, acc[0][j]);
        acc[1][j] = wmma_f32_f16(a1, bf, acc[1][j]);
      }
    }
    __syncthreads();          // reads done before buffer is re-filled
  }

#pragma unroll
  for (int mi = 0; mi < 2; ++mi) {
#pragma unroll
    for (int j = 0; j < 4; ++j) {
      int n = n0 + j * 16 + nl;
      if (n < N) {
        float bn = bias ? bias[n] : 0.f;
#pragma unroll
        for (int v = 0; v < 8; ++v) {
          float val = acc[mi][j][v] + bn;
          if (ACT == 1) val = (val > 20.f) ? val : log1pf(__expf(val));  // softplus
          size_t m = (size_t)(m0 + mi * 16 + 8 * hi + v);
          D[m * ldd + n] = val;
          if (WRITE_F16) D16[m * ldd16 + n] = (_Float16)val;
        }
      }
    }
  }
}

// ---------------- depthwise causal conv (k=4) + SiLU ----------------
__global__ void dwconv_silu_kernel(const float* __restrict__ xz,  // (MFULL, 2*DIN), x_in = cols [0,DIN)
                                   const float* __restrict__ cw,  // (DIN,1,4)
                                   const float* __restrict__ cb,
                                   float* __restrict__ xs,
                                   _Float16* __restrict__ xs16) {
  size_t i = (size_t)blockIdx.x * blockDim.x + threadIdx.x;
  if (i >= (size_t)MFULL * DIN) return;
  int c = (int)(i & (DIN - 1));
  size_t row = i >> 10;                 // b*TT + t
  int t = (int)(row & (TT - 1));
  float w0 = cw[c * 4 + 0], w1 = cw[c * 4 + 1], w2 = cw[c * 4 + 2], w3 = cw[c * 4 + 3];
  const float* base = xz + row * (2 * DIN) + c;
  float acc = cb[c] + base[0] * w3;
  if (t >= 1) acc += base[-1 * 2 * DIN] * w2;
  if (t >= 2) acc += base[-2 * 2 * DIN] * w1;
  if (t >= 3) acc += base[-3 * 2 * DIN] * w0;
  float s = acc / (1.f + __expf(-acc));        // SiLU
  xs[i]   = s;
  xs16[i] = (_Float16)s;
}

// ---------------- selective scan + D-residual + SiLU(z) gate ----------------
__global__ void ssm_scan_kernel(const float* __restrict__ dt,    // (MFULL, DIN)
                                const float* __restrict__ xdbl,  // (MFULL, 80): [dtr|B|C]
                                const float* __restrict__ xs,    // (MFULL, DIN)
                                const float* __restrict__ xz,    // (MFULL, 2*DIN): z = cols [DIN,2DIN)
                                const float* __restrict__ A_log, // (DIN, NS)
                                const float* __restrict__ Dp,    // (DIN)
                                _Float16* __restrict__ yg) {     // (MFULL, DIN)
  int c = blockIdx.x * blockDim.x + threadIdx.x;   // channel
  int b = blockIdx.y;
  float Ac[NS];
#pragma unroll
  for (int s = 0; s < NS; ++s) Ac[s] = -__expf(A_log[(size_t)c * NS + s]);
  float hst[NS];
#pragma unroll
  for (int s = 0; s < NS; ++s) hst[s] = 0.f;
  float dpc = Dp[c];

  for (int t = 0; t < TT; ++t) {
    size_t row = (size_t)b * TT + t;
    float dtv = dt[row * DIN + c];
    float xv  = xs[row * DIN + c];
    float zv  = xz[row * (2 * DIN) + DIN + c];
    const float* bm = xdbl + row * NXP + DTR;
    const float* cm = xdbl + row * NXP + DTR + NS;
    float dx = dtv * xv;
    float y = 0.f;
#pragma unroll
    for (int s = 0; s < NS; ++s) {
      float da = __expf(dtv * Ac[s]);
      hst[s] = hst[s] * da + dx * bm[s];
      y += hst[s] * cm[s];
    }
    y += xv * dpc;
    y *= zv / (1.f + __expf(-zv));               // * SiLU(z)
    yg[row * DIN + c] = (_Float16)y;
  }
}

// ---------------- host-side orchestration ----------------
extern "C" void kernel_launch(void* const* d_in, const int* in_sizes, int n_in,
                              void* d_out, int out_size, void* d_ws, size_t ws_size,
                              hipStream_t stream) {
  const float* x      = (const float*)d_in[0];
  const float* skip   = (const float*)d_in[1];
  const float* up_w   = (const float*)d_in[2];
  const float* up_b   = (const float*)d_in[3];
  const float* fus_w  = (const float*)d_in[4];
  const float* fus_b  = (const float*)d_in[5];
  const float* ln_g   = (const float*)d_in[6];
  const float* ln_b   = (const float*)d_in[7];
  const float* in_w   = (const float*)d_in[8];
  const float* conv_w = (const float*)d_in[9];
  const float* conv_b = (const float*)d_in[10];
  const float* xproj_w= (const float*)d_in[11];
  const float* dt_w   = (const float*)d_in[12];
  const float* dt_b   = (const float*)d_in[13];
  const float* A_log  = (const float*)d_in[14];
  const float* Dp     = (const float*)d_in[15];
  const float* out_w  = (const float*)d_in[16];
  float* out = (float*)d_out;

  // workspace carve-out (256B aligned regions)
  size_t off = 0;
  char* ws = (char*)d_ws;
  auto take = [&](size_t bytes) -> char* {
    char* p = ws + off;
    off += (bytes + 255) & ~(size_t)255;
    return p;
  };
  _Float16* x16    = (_Float16*)take((size_t)NB * TLOW * DM * 2);
  _Float16* sk16   = (_Float16*)take((size_t)NB * TT * DM * 2);
  _Float16* Ct     = (_Float16*)take((size_t)4 * DM * DM * 2);
  _Float16* Fsw    = (_Float16*)take((size_t)DM * DM * 2);
  float*    bias2  = (float*)   take((size_t)DM * 4);
  _Float16* inw16  = (_Float16*)take((size_t)2 * DIN * DM * 2);
  _Float16* xpw16  = (_Float16*)take((size_t)NXP * DIN * 2);
  _Float16* dtw16  = (_Float16*)take((size_t)DIN * DTR * 2);
  _Float16* outw16 = (_Float16*)take((size_t)DM * DIN * 2);
  float*    fused  = (float*)   take((size_t)MFULL * DM * 4);
  _Float16* h16    = (_Float16*)take((size_t)MFULL * DM * 2);
  float*    xzbuf  = (float*)   take((size_t)MFULL * 2 * DIN * 4);
  float*    xsf    = (float*)   take((size_t)MFULL * DIN * 4);
  _Float16* xs16   = (_Float16*)take((size_t)MFULL * DIN * 2);
  float*    xdbl   = (float*)   take((size_t)MFULL * NXP * 4);
  _Float16* xdbl16 = (_Float16*)take((size_t)MFULL * NXP * 2);
  float*    dtf    = (float*)   take((size_t)MFULL * DIN * 4);
  _Float16* yg16   = (_Float16*)take((size_t)MFULL * DIN * 2);
  (void)ws_size; (void)n_in; (void)in_sizes; (void)out_size;

  auto cvt = [&](const float* s, _Float16* d, size_t n) {
    cvt_f16_kernel<<<dim3((unsigned)((n + 255) / 256)), dim3(256), 0, stream>>>(s, d, n);
  };
  // f32 -> f16 operand staging
  cvt(x,       x16,    (size_t)NB * TLOW * DM);
  cvt(skip,    sk16,   (size_t)NB * TT * DM);
  cvt(in_w,    inw16,  (size_t)2 * DIN * DM);
  cvt(xproj_w, xpw16,  (size_t)NXP * DIN);
  cvt(dt_w,    dtw16,  (size_t)DIN * DTR);
  cvt(out_w,   outw16, (size_t)DM * DIN);

  // composite convT*fusion weights + fused bias + skip weights
  build_composite_kernel<<<dim3((4 * DM * DM + 255) / 256), dim3(256), 0, stream>>>(up_w, fus_w, Ct);
  build_bias2_kernel<<<dim3((DM + 255) / 256), dim3(256), 0, stream>>>(up_b, fus_w, fus_b, bias2);
  extract_fs_kernel<<<dim3((DM * DM + 255) / 256), dim3(256), 0, stream>>>(fus_w, Fsw);

  // fused upsample + skip-fusion (even/odd output rows)
  {
    dim3 grid(DM / 64, NB * (TLOW / 16));
    fuse_kernel<<<grid, dim3(128), 0, stream>>>(x16, sk16, Ct, Fsw, bias2, fused, 0);
    fuse_kernel<<<grid, dim3(128), 0, stream>>>(x16, sk16, Ct, Fsw, bias2, fused, 1);
  }

  // LayerNorm -> h (f16)
  ln_kernel<<<dim3(MFULL), dim3(256), 0, stream>>>(fused, ln_g, ln_b, h16);

  // in_proj: xz = h @ in_w^T   (32768 x 2048 x 512)
  wmma_gemm_kernel<0, false><<<dim3(2 * DIN / 64, MFULL / 128), dim3(128), 0, stream>>>(
      h16, DM, inw16, DM, nullptr, xzbuf, 2 * DIN, nullptr, 0, MFULL, 2 * DIN, DM);

  // depthwise causal conv + SiLU
  dwconv_silu_kernel<<<dim3((unsigned)(((size_t)MFULL * DIN + 255) / 256)), dim3(256), 0, stream>>>(
      xzbuf, conv_w, conv_b, xsf, xs16);

  // x_proj: x_dbl = xs @ xproj_w^T  (32768 x 80 x 1024), keep f32 + f16
  wmma_gemm_kernel<0, true><<<dim3((NXP + 63) / 64, MFULL / 128), dim3(128), 0, stream>>>(
      xs16, DIN, xpw16, DIN, nullptr, xdbl, NXP, xdbl16, NXP, MFULL, NXP, DIN);

  // dt = softplus(dtr @ dt_w^T + dt_b)  (32768 x 1024 x 32)
  wmma_gemm_kernel<1, false><<<dim3(DIN / 64, MFULL / 128), dim3(128), 0, stream>>>(
      xdbl16, NXP, dtw16, DTR, dt_b, dtf, DIN, nullptr, 0, MFULL, DIN, DTR);

  // selective scan + gate -> yg (f16)
  ssm_scan_kernel<<<dim3(DIN / 256, NB), dim3(256), 0, stream>>>(
      dtf, xdbl, xsf, xzbuf, A_log, Dp, yg16);

  // out_proj: out = yg @ out_w^T  (32768 x 512 x 1024)
  wmma_gemm_kernel<0, false><<<dim3(DM / 64, MFULL / 128), dim3(128), 0, stream>>>(
      yg16, DIN, outw16, DIN, nullptr, out, DM, nullptr, 0, MFULL, DM, DIN);
}
